// MultiHeadAttention_18674517803234
// MI455X (gfx1250) — matrix operand order repaired and checked by reading, not compiled, not run on hardware
//
#include <hip/hip_runtime.h>
#include <hip/hip_bf16.h>

// ---------------------------------------------------------------------------
// MHA forward for MI455X (gfx1250): f16 WMMA (fp32 accum) everywhere.
// B=4, S=2048, D=1024, H=16, Hd=64, M = B*S = 8192.
// ---------------------------------------------------------------------------

typedef __attribute__((ext_vector_type(16))) _Float16 v16h;
typedef __attribute__((ext_vector_type(8)))  _Float16 v8h;
typedef __attribute__((ext_vector_type(4)))  _Float16 v4h;
typedef __attribute__((ext_vector_type(8)))  float    v8f;
typedef __attribute__((ext_vector_type(4)))  unsigned int u32x4;
typedef __attribute__((ext_vector_type(8)))  int      i32x8;
typedef __attribute__((ext_vector_type(4)))  int      i32x4;

#define DIM      1024
#define SEQ      2048
#define NHEADS   16
#define HEADDIM  64
#define MTOK     8192          // B*S
#define ATTSCALE 0.125f        // 1/sqrt(64)

#if __has_builtin(__builtin_amdgcn_tensor_load_to_lds) && \
    __has_builtin(__builtin_amdgcn_s_wait_tensorcnt)
#define USE_TDM 1
#else
#define USE_TDM 0
#endif

#if __has_builtin(__builtin_amdgcn_update_dpp)
#define USE_DPP 1
#else
#define USE_DPP 0
#endif

// ---------------- WMMA operand loaders (CDNA5 16x16x32 f16 layouts) --------

// A matrix 16x32 (MxK), row-major source with leading dim ld (halves).
// Lane L: row = L&15; halves 0..15 -> K = (L>>4)*16 + i.
__device__ __forceinline__ v16h load_a(const _Float16* p, int ld) {
  const int lane = threadIdx.x & 31;
  const int row  = lane & 15;
  const int khi  = (lane >> 4) * 16;
  const _Float16* r = p + row * ld + khi;
  v8h lo = *(const v8h*)(r);
  v8h hi = *(const v8h*)(r + 8);
  v16h a;
#pragma unroll
  for (int i = 0; i < 8; ++i) { a[i] = lo[i]; a[i + 8] = hi[i]; }
  return a;
}

// B matrix 32x16 (KxN) where B[k][n] = src[n*ld + k]  (lane reads a row chunk:
// used for x @ W^T and Q @ K^T, both of which want rows of the source).
// Lane L: n = L&15; k = (L>>4)*16 + i.
__device__ __forceinline__ v16h load_b_row(const _Float16* p, int ld) {
  const int lane = threadIdx.x & 31;
  const int n  = lane & 15;
  const int k0 = (lane >> 4) * 16;
  const _Float16* r = p + n * ld + k0;
  v8h lo = *(const v8h*)(r);
  v8h hi = *(const v8h*)(r + 8);
  v16h b;
#pragma unroll
  for (int i = 0; i < 8; ++i) { b[i] = lo[i]; b[i + 8] = hi[i]; }
  return b;
}

// B matrix 32x16 (KxN) where B[k][n] = src[k*ld + n]  (column gather: P @ V).
__device__ __forceinline__ v16h load_b_col(const _Float16* p, int ld) {
  const int lane = threadIdx.x & 31;
  const int n  = lane & 15;
  const int k0 = (lane >> 4) * 16;
  v16h b;
#pragma unroll
  for (int i = 0; i < 16; ++i) b[i] = p[(k0 + i) * ld + n];
  return b;
}

__device__ __forceinline__ v8f wmma_f16(v16h a, v16h b, v8f c) {
  return __builtin_amdgcn_wmma_f32_16x16x32_f16(false, a, false, b, (short)0, c,
                                                false, false);
}

// ---------------- 16-lane row reductions via DPP16 ROW_ROR -----------------
// Reduction stays within each 16-lane half-wave, matching the WMMA C-matrix
// row grouping (lanes 0-15 = rows M=0..7, lanes 16-31 = rows M=8..15).
// ROW_ROR:n ctrl = 0x120+n; VALU-only, no memory counters / waits.

#if USE_DPP
template <int N>
__device__ __forceinline__ float dpp_ror_f32(float x) {
  const int xi = __float_as_int(x);
  return __int_as_float(
      __builtin_amdgcn_update_dpp(xi, xi, 0x120 + N, 0xF, 0xF, true));
}
__device__ __forceinline__ float row16_max(float x) {
  x = fmaxf(x, dpp_ror_f32<8>(x));
  x = fmaxf(x, dpp_ror_f32<4>(x));
  x = fmaxf(x, dpp_ror_f32<2>(x));
  x = fmaxf(x, dpp_ror_f32<1>(x));
  return x;
}
__device__ __forceinline__ float row16_sum(float x) {
  x += dpp_ror_f32<8>(x);
  x += dpp_ror_f32<4>(x);
  x += dpp_ror_f32<2>(x);
  x += dpp_ror_f32<1>(x);
  return x;
}
#else
__device__ __forceinline__ float row16_max(float x) {
#pragma unroll
  for (int ofs = 1; ofs < 16; ofs <<= 1) x = fmaxf(x, __shfl_xor(x, ofs, 32));
  return x;
}
__device__ __forceinline__ float row16_sum(float x) {
#pragma unroll
  for (int ofs = 1; ofs < 16; ofs <<= 1) x += __shfl_xor(x, ofs, 32);
  return x;
}
#endif

// ---------------- Tensor Data Mover: 2D f16 tile -> LDS --------------------
// Builds D# group0/group1 per CDNA5 ISA §8.3/8.4 (2D tensor, data_size=2B).
// This toolchain exposes the 6-arg builtin:
//   (uint32x4 g0, int32x8 g1, int32x4 g2, int32x4 g3, int32x8 g4, i32 cpol)
__device__ __forceinline__ void tdm_load_tile_2d(const _Float16* gbase,
                                                 unsigned lds_byte,
                                                 unsigned tile_rows,
                                                 unsigned tile_cols,
                                                 unsigned tensor_rows,
                                                 unsigned row_stride_elems) {
#if USE_TDM
  const unsigned long long ga = (unsigned long long)(size_t)gbase;
  u32x4 g0;
  g0[0] = 1u;                                     // count=1 (valid), user mode
  g0[1] = lds_byte;                               // lds_addr (bytes)
  g0[2] = (unsigned)ga;                           // global_addr[31:0]
  g0[3] = (unsigned)((ga >> 32) & 0x1FFFFFFu)     // global_addr[56:32]
          | 0x80000000u;                          // type=2 ("image")
  const unsigned td0 = tile_cols;                 // tensor_dim0 (elems/row)
  const unsigned td1 = tensor_rows;               // tensor_dim1
  const unsigned st0 = row_stride_elems;          // tensor_dim0_stride
  i32x8 g1;
  g1[0] = (int)(1u << 16);                        // workgroup_mask=0, data_size=1 (2B)
  g1[1] = (int)((td0 & 0xFFFFu) << 16);           // abar_addr=0 | dim0.lo16
  g1[2] = (int)(((td0 >> 16) & 0xFFFFu) | ((td1 & 0xFFFFu) << 16));
  g1[3] = (int)(((td1 >> 16) & 0xFFFFu) | ((tile_cols & 0xFFFFu) << 16)); // tile_dim0
  g1[4] = (int)(tile_rows & 0xFFFFu);             // tile_dim1 | tile_dim2=0
  g1[5] = (int)(st0);                             // dim0_stride lo32
  g1[6] = 0;                                      // dim0_stride hi16 | dim1_stride lo16
  g1[7] = 0;                                      // dim1_stride hi32
  const i32x4 z4 = {0, 0, 0, 0};
  const i32x8 z8 = {0, 0, 0, 0, 0, 0, 0, 0};
  __builtin_amdgcn_tensor_load_to_lds(g0, g1, z4, z4, z8, 0);
#else
  (void)gbase; (void)lds_byte; (void)tile_rows; (void)tile_cols;
  (void)tensor_rows; (void)row_stride_elems;
#endif
}

// ---------------- 1) fused LayerNorm + cast to f16 -------------------------

__global__ void ln_cast_kernel(const float* __restrict__ x,
                               const float* __restrict__ gamma,
                               const float* __restrict__ beta,
                               _Float16* __restrict__ xn) {
  const int row = blockIdx.x;           // 0..8191
  const int tid = threadIdx.x;          // 256 threads, 4 elems each
  const float4 v = ((const float4*)(x + (size_t)row * DIM))[tid];
  float s  = v.x + v.y + v.z + v.w;
  float s2 = v.x * v.x + v.y * v.y + v.z * v.z + v.w * v.w;

  __shared__ float rs[256];
  __shared__ float rq[256];
  rs[tid] = s; rq[tid] = s2;
  __syncthreads();
#pragma unroll
  for (int off = 128; off > 0; off >>= 1) {
    if (tid < off) { rs[tid] += rs[tid + off]; rq[tid] += rq[tid + off]; }
    __syncthreads();
  }
  const float mu  = rs[0] * (1.0f / DIM);
  const float var = rq[0] * (1.0f / DIM) - mu * mu;
  const float inv = rsqrtf(var + 1e-5f);

  const float4 g4 = ((const float4*)gamma)[tid];
  const float4 b4 = ((const float4*)beta)[tid];
  v4h o;
  o[0] = (_Float16)((v.x - mu) * inv * g4.x + b4.x);
  o[1] = (_Float16)((v.y - mu) * inv * g4.y + b4.y);
  o[2] = (_Float16)((v.z - mu) * inv * g4.z + b4.z);
  o[3] = (_Float16)((v.w - mu) * inv * g4.w + b4.w);
  ((v4h*)(xn + (size_t)row * DIM))[tid] = o;
}

// ---------------- 2) weight cast f32 -> f16 --------------------------------

__global__ void cast_w_kernel(const float* __restrict__ w,
                              _Float16* __restrict__ wh) {
  const int i = (blockIdx.x * 256 + threadIdx.x) * 4;
  const float4 v = *(const float4*)(w + i);
  v4h o;
  o[0] = (_Float16)v.x; o[1] = (_Float16)v.y;
  o[2] = (_Float16)v.z; o[3] = (_Float16)v.w;
  *(v4h*)(wh + i) = o;
}

// ---------------- 3) GEMM:  C[M,N] = A[M,K] @ W[N,K]^T + bias --------------
// Each wave: 32x64 output tile (2 A operands, 8 fp32 accumulators; each B
// operand feeds two WMMAs). Block: 8 waves = 64 rows x 256 cols.
// MODE 0: fp32 row-major output [MTOK, DIM]
// MODE 1: f16 output scattered to [B, H, S, Hd]

template <int MODE>
__device__ __forceinline__ void gemm_store(const v8f acc[4], int m0, int n0,
                                           const float* __restrict__ bias,
                                           void* __restrict__ outp, int lane) {
  const int mbase = (lane >> 4) * 8;
  const int ncol  = lane & 15;
#pragma unroll
  for (int j = 0; j < 4; ++j) {
    const int n  = n0 + j * 16 + ncol;
    const float bv = bias[n];
#pragma unroll
    for (int r = 0; r < 8; ++r) {
      const int m = m0 + mbase + r;
      const float val = acc[j][r] + bv;
      if (MODE == 0) {
        ((float*)outp)[(size_t)m * DIM + n] = val;
      } else {
        const int b = m >> 11, s = m & (SEQ - 1);
        const int h = n >> 6, d = n & (HEADDIM - 1);
        ((_Float16*)outp)[(((size_t)(b * NHEADS + h)) * SEQ + s) * HEADDIM + d] =
            (_Float16)val;
      }
    }
  }
}

template <int MODE>
__global__ __launch_bounds__(256, 1)
void gemm_kernel(const _Float16* __restrict__ A,
                 const _Float16* __restrict__ W,
                 const float* __restrict__ bias,
                 void* __restrict__ outp) {
  const int lane = threadIdx.x & 31;
  const int wv   = threadIdx.x >> 5;                   // 0..7
  const int m0   = blockIdx.x * 64 + (wv & 1) * 32;    // 64 rows per block
  const int n0   = blockIdx.y * 256 + (wv >> 1) * 64;  // 256 cols per block

  v8f acc0[4] = {};
  v8f acc1[4] = {};
#pragma unroll 2
  for (int kt = 0; kt < DIM; kt += 32) {
    const _Float16* ap = A + (size_t)m0 * DIM + kt;
    if (kt + 32 < DIM) {
      __builtin_prefetch(ap + 32, 0, 3);               // global_prefetch_b8
      __builtin_prefetch(W + (size_t)n0 * DIM + kt + 32, 0, 3);
    }
    const v16h a0 = load_a(ap, DIM);
    const v16h a1 = load_a(ap + 16 * DIM, DIM);
#pragma unroll
    for (int j = 0; j < 4; ++j) {
      const v16h b = load_b_row(W + (size_t)(n0 + j * 16) * DIM + kt, DIM);
      acc0[j] = wmma_f16(a0, b, acc0[j]);
      acc1[j] = wmma_f16(a1, b, acc1[j]);
    }
  }
  gemm_store<MODE>(acc0, m0,      n0, bias, outp, lane);
  gemm_store<MODE>(acc1, m0 + 16, n0, bias, outp, lane);
}

// ---------------- 4) flash attention per (b,h) -----------------------------
// grid.x = B*H (64), grid.y = S/128 (16); 8 waves, 16 query rows per wave.
__global__ __launch_bounds__(256, 1)
void attn_kernel(const _Float16* __restrict__ Q,
                 const _Float16* __restrict__ K,
                 const _Float16* __restrict__ V,
                 _Float16* __restrict__ O) {
  __shared__ _Float16 ldsK[64 * HEADDIM];   // 8 KB
  __shared__ _Float16 ldsV[64 * HEADDIM];   // 8 KB
  __shared__ _Float16 ldsP[8 * 16 * 64];    // 16 KB, wave-private tiles

  const int tid  = threadIdx.x;
  const int lane = tid & 31;
  const int wv   = tid >> 5;
  const int bh   = blockIdx.x;
  const int b    = bh >> 4;
  const int h    = bh & (NHEADS - 1);
  const int q0   = blockIdx.y * 128 + wv * 16;

  const _Float16* Qh = Q + (size_t)bh * SEQ * HEADDIM;
  const _Float16* Kh = K + (size_t)bh * SEQ * HEADDIM;
  const _Float16* Vh = V + (size_t)bh * SEQ * HEADDIM;

  // Q tile (16 x 64), pre-scaled by 1/sqrt(Hd) (0.125 exact in f16)
  v16h aq0 = load_a(Qh + (size_t)q0 * HEADDIM + 0, HEADDIM);
  v16h aq1 = load_a(Qh + (size_t)q0 * HEADDIM + 32, HEADDIM);
#pragma unroll
  for (int i = 0; i < 16; ++i) {
    aq0[i] = aq0[i] * (_Float16)ATTSCALE;
    aq1[i] = aq1[i] * (_Float16)ATTSCALE;
  }

  v8f oacc[4] = {};
  float m8[8], l8[8];
#pragma unroll
  for (int r = 0; r < 8; ++r) { m8[r] = -1e30f; l8[r] = 0.0f; }

  _Float16* Pw = ldsP + wv * 16 * 64;
#if USE_TDM
  const unsigned ldsK_off = (unsigned)(size_t)(&ldsK[0]);
  const unsigned ldsV_off = (unsigned)(size_t)(&ldsV[0]);
#endif

  for (int kt = 0; kt < SEQ; kt += 64) {
    __syncthreads();  // previous chunk fully consumed by all waves
#if USE_TDM
    if (wv == 0) {
      tdm_load_tile_2d(Kh + (size_t)kt * HEADDIM, ldsK_off, 64, HEADDIM, SEQ,
                       HEADDIM);
      tdm_load_tile_2d(Vh + (size_t)kt * HEADDIM, ldsV_off, 64, HEADDIM, SEQ,
                       HEADDIM);
      __builtin_amdgcn_s_wait_tensorcnt(0);
    }
#else
    {
      const v8h* gK = (const v8h*)(Kh + (size_t)kt * HEADDIM);
      const v8h* gV = (const v8h*)(Vh + (size_t)kt * HEADDIM);
      v8h* sK = (v8h*)ldsK;
      v8h* sV = (v8h*)ldsV;
#pragma unroll
      for (int i = 0; i < 2; ++i) {      // 512 v8h per array / 256 threads
        sK[tid + i * 256] = gK[tid + i * 256];
        sV[tid + i * 256] = gV[tid + i * 256];
      }
    }
#endif
    __syncthreads();

    // ---- S = Q K^T for 64 keys: 4 chunks of 16 keys ----
    v8f sc[4] = {};
#pragma unroll
    for (int j = 0; j < 4; ++j) {
      const v16h bk0 = load_b_row(ldsK + (j * 16) * HEADDIM + 0, HEADDIM);
      sc[j] = wmma_f16(aq0, bk0, sc[j]);
      const v16h bk1 = load_b_row(ldsK + (j * 16) * HEADDIM + 32, HEADDIM);
      sc[j] = wmma_f16(aq1, bk1, sc[j]);
    }

    // ---- online softmax update; row reductions via DPP (no LDS waits) ----
    float corr[8];
#pragma unroll
    for (int r = 0; r < 8; ++r) {
      float v = fmaxf(fmaxf(sc[0][r], sc[1][r]), fmaxf(sc[2][r], sc[3][r]));
      v = row16_max(v);
      const float mnew = fmaxf(m8[r], v);
      corr[r] = __expf(m8[r] - mnew);
      m8[r] = mnew;
      float ps = 0.0f;
#pragma unroll
      for (int j = 0; j < 4; ++j) {
        const float p = __expf(sc[j][r] - mnew);
        sc[j][r] = p;
        ps += p;
      }
      l8[r] = l8[r] * corr[r] + row16_sum(ps);
    }
#pragma unroll
    for (int j = 0; j < 4; ++j)
#pragma unroll
      for (int r = 0; r < 8; ++r) oacc[j][r] *= corr[r];

    // ---- transpose P (C layout -> A layout) through wave-private LDS ----
    const int mbase = (lane >> 4) * 8;
    const int ncol  = lane & 15;
#pragma unroll
    for (int j = 0; j < 4; ++j)
#pragma unroll
      for (int r = 0; r < 8; ++r)
        Pw[(mbase + r) * 64 + j * 16 + ncol] = (_Float16)sc[j][r];

    const v16h ap0 = load_a(Pw + 0, 64);
    const v16h ap1 = load_a(Pw + 32, 64);

    // ---- O += P V ----
#pragma unroll
    for (int j = 0; j < 4; ++j) {
      const v16h bv0 = load_b_col(ldsV + 0 * HEADDIM + j * 16, HEADDIM);
      oacc[j] = wmma_f16(ap0, bv0, oacc[j]);
      const v16h bv1 = load_b_col(ldsV + 32 * HEADDIM + j * 16, HEADDIM);
      oacc[j] = wmma_f16(ap1, bv1, oacc[j]);
    }
  }

  // ---- normalize and store as [B*S, D] f16 for the O-projection GEMM ----
  const int mbase = (lane >> 4) * 8;
  const int ncol  = lane & 15;
#pragma unroll
  for (int j = 0; j < 4; ++j) {
#pragma unroll
    for (int r = 0; r < 8; ++r) {
      const int s   = q0 + mbase + r;
      const int col = h * HEADDIM + j * 16 + ncol;
      O[((size_t)(b * SEQ + s)) * DIM + col] =
          (_Float16)(oacc[j][r] / l8[r]);
    }
  }
}

// ---------------------------------------------------------------------------

extern "C" void kernel_launch(void* const* d_in, const int* in_sizes, int n_in,
                              void* d_out, int out_size, void* d_ws, size_t ws_size,
                              hipStream_t stream) {
  (void)in_sizes; (void)n_in; (void)out_size; (void)ws_size;
  const float* x     = (const float*)d_in[0];
  const float* gamma = (const float*)d_in[1];
  const float* beta  = (const float*)d_in[2];
  const float* Wq    = (const float*)d_in[3];
  const float* bq    = (const float*)d_in[4];
  const float* Wk    = (const float*)d_in[5];
  const float* bk    = (const float*)d_in[6];
  const float* Wv    = (const float*)d_in[7];
  const float* bv    = (const float*)d_in[8];
  const float* Wo    = (const float*)d_in[9];
  const float* bo    = (const float*)d_in[10];

  _Float16* ws  = (_Float16*)d_ws;
  _Float16* xn  = ws;                            // 8M halves
  _Float16* Wqh = xn  + (size_t)MTOK * DIM;      // 1M each
  _Float16* Wkh = Wqh + (size_t)DIM * DIM;
  _Float16* Wvh = Wkh + (size_t)DIM * DIM;
  _Float16* Woh = Wvh + (size_t)DIM * DIM;
  _Float16* Qh  = Woh + (size_t)DIM * DIM;       // 8M each, [B,H,S,Hd]
  _Float16* Kh  = Qh  + (size_t)MTOK * DIM;
  _Float16* Vh  = Kh  + (size_t)MTOK * DIM;
  _Float16* AO  = Vh  + (size_t)MTOK * DIM;      // 8M, [B*S, D]

  ln_cast_kernel<<<MTOK, 256, 0, stream>>>(x, gamma, beta, xn);

  cast_w_kernel<<<DIM * DIM / 1024, 256, 0, stream>>>(Wq, Wqh);
  cast_w_kernel<<<DIM * DIM / 1024, 256, 0, stream>>>(Wk, Wkh);
  cast_w_kernel<<<DIM * DIM / 1024, 256, 0, stream>>>(Wv, Wvh);
  cast_w_kernel<<<DIM * DIM / 1024, 256, 0, stream>>>(Wo, Woh);

  const dim3 gg(MTOK / 64, DIM / 256);
  gemm_kernel<1><<<gg, 256, 0, stream>>>(xn, Wqh, bq, Qh);
  gemm_kernel<1><<<gg, 256, 0, stream>>>(xn, Wkh, bk, Kh);
  gemm_kernel<1><<<gg, 256, 0, stream>>>(xn, Wvh, bv, Vh);

  attn_kernel<<<dim3(4 * NHEADS, SEQ / 128), 256, 0, stream>>>(Qh, Kh, Vh, AO);

  gemm_kernel<0><<<gg, 256, 0, stream>>>(AO, Woh, bo, d_out);
}
